// QDense_81621558493457
// MI455X (gfx1250) — 8-bit matrix operand order repaired and checked by reading, compiled, not run on hardware
//
#include <hip/hip_runtime.h>

typedef int v8i __attribute__((ext_vector_type(8)));

#define QMAX 127.0f

__global__ __launch_bounds__(256) void quant_x_kernel(
    const float* __restrict__ x, signed char* __restrict__ qxA,
    float* __restrict__ sx) {
  __shared__ float red[256];
  const int row = blockIdx.x;
  const int t = threadIdx.x;
  const float* xr = x + (size_t)row * 2048;

  float v[8];
  float m = 0.0f;
#pragma unroll
  for (int i = 0; i < 8; ++i) {
    v[i] = xr[t * 8 + i];
    m = fmaxf(m, fabsf(v[i]));
  }
  red[t] = m;
  __syncthreads();
  for (int s = 128; s > 0; s >>= 1) {
    if (t < s) red[t] = fmaxf(red[t], red[t + s]);
    __syncthreads();
  }
  const float amax = red[0];
  const float scale = amax > 0.0f ? amax * (1.0f / QMAX) : 1.0f;
  if (t == 0) sx[row] = scale;

  const int mt = row >> 4, ml = row & 15;
  signed char* tile0 = qxA + (size_t)mt * 32 * 1024;
#pragma unroll
  for (int i = 0; i < 8; ++i) {
    const int k = t * 8 + i;
    const int kt = k >> 6, ko = k & 63;
    const int lane = ml + (((ko >> 3) & 1) << 4);
    const int byt = ((ko >> 4) << 3) + (ko & 7);
    float q = rintf(v[i] / scale);
    q = fminf(fmaxf(q, -QMAX), QMAX);
    tile0[(size_t)kt * 1024 + lane * 32 + byt] = (signed char)(int)q;
  }
}

__global__ __launch_bounds__(256) void quant_w_kernel(
    const float* __restrict__ kern, signed char* __restrict__ qkB,
    float* __restrict__ sk) {
  __shared__ float red[256];
  const int t = threadIdx.x;
  const int c = t & 63, rr = t >> 6;
  const int col = blockIdx.x * 64 + c;

  float m = 0.0f;
  for (int d = rr; d < 2048; d += 4)
    m = fmaxf(m, fabsf(kern[(size_t)d * 2048 + col]));
  red[t] = m;
  __syncthreads();
  if (rr == 0)
    red[c] = fmaxf(fmaxf(red[c], red[c + 64]), fmaxf(red[c + 128], red[c + 192]));
  __syncthreads();
  const float amax = red[c];
  const float scale = amax > 0.0f ? amax * (1.0f / QMAX) : 1.0f;
  if (rr == 0) sk[col] = scale;

  const int nt = col >> 4, n = col & 15;
  signed char* tile0 = qkB + (size_t)nt * 32 * 1024;
  for (int d = rr; d < 2048; d += 4) {
    float q = rintf(kern[(size_t)d * 2048 + col] / scale);
    q = fminf(fmaxf(q, -QMAX), QMAX);
    const int kt = d >> 6, ko = d & 63;
    const int lane = n + (((ko >> 3) & 1) << 4);
    const int byt = ((ko >> 4) << 3) + (ko & 7);
    tile0[(size_t)kt * 1024 + lane * 32 + byt] = (signed char)(int)q;
  }
}

__global__ __launch_bounds__(256) void quant_b_kernel(
    const float* __restrict__ bias, float* __restrict__ bq) {
  __shared__ float red[256];
  const int t = threadIdx.x;
  float v[8];
  float m = 0.0f;
#pragma unroll
  for (int i = 0; i < 8; ++i) {
    v[i] = bias[t * 8 + i];
    m = fmaxf(m, fabsf(v[i]));
  }
  red[t] = m;
  __syncthreads();
  for (int s = 128; s > 0; s >>= 1) {
    if (t < s) red[t] = fmaxf(red[t], red[t + s]);
    __syncthreads();
  }
  const float amax = red[0];
  const float scale = amax > 0.0f ? amax * (1.0f / QMAX) : 1.0f;
#pragma unroll
  for (int i = 0; i < 8; ++i) {
    float q = rintf(v[i] / scale);
    q = fminf(fmaxf(q, -QMAX), QMAX);
    bq[t * 8 + i] = q * scale;
  }
}

__global__ __launch_bounds__(256) void qgemm_wmma_kernel(
    const signed char* __restrict__ qxA, const float* __restrict__ sx,
    const signed char* __restrict__ qkB, const float* __restrict__ sk,
    const float* __restrict__ bq, float* __restrict__ out) {
  const int lane = threadIdx.x & 31;
  const int w = threadIdx.x >> 5;
  const int mt = blockIdx.y * 2 + (w >> 2);
  const int nt0 = blockIdx.x * 8 + ((w & 3) << 1);
  const int nt1 = nt0 + 1;

  const signed char* aBase = qxA + (size_t)mt * 32 * 1024 + lane * 32;
  const signed char* b0Base = qkB + (size_t)nt0 * 32 * 1024 + lane * 32;
  const signed char* b1Base = qkB + (size_t)nt1 * 32 * 1024 + lane * 32;

  v8i acc0 = {};
  v8i acc1 = {};
#pragma unroll 4
  for (int kt = 0; kt < 32; ++kt) {
    const v8i a = *(const v8i*)(aBase + (size_t)kt * 1024);
    const v8i b0 = *(const v8i*)(b0Base + (size_t)kt * 1024);
    const v8i b1 = *(const v8i*)(b1Base + (size_t)kt * 1024);
    acc0 = __builtin_amdgcn_wmma_i32_16x16x64_iu8(true, a, true, b0, acc0,
                                                  false, false);
    acc1 = __builtin_amdgcn_wmma_i32_16x16x64_iu8(true, a, true, b1, acc1,
                                                  false, false);
  }

  const int mrow0 = mt * 16 + ((lane >> 4) << 3);
  const int ncol = lane & 15;
  const int col0 = nt0 * 16 + ncol;
  const int col1 = nt1 * 16 + ncol;
  const float sk0 = sk[col0], sk1 = sk[col1];
  const float bq0 = bq[col0], bq1 = bq[col1];
#pragma unroll
  for (int r = 0; r < 8; ++r) {
    const int row = mrow0 + r;
    const float srow = sx[row];
    float y0 = (float)acc0[r] * srow * sk0 + bq0;
    float y1 = (float)acc1[r] * srow * sk1 + bq1;
    y0 = fmaxf(y0, 0.0f);
    y1 = fmaxf(y1, 0.0f);
    out[(size_t)row * 2048 + col0] = y0;
    out[(size_t)row * 2048 + col1] = y1;
  }
}

__global__ __launch_bounds__(256) void requant_out_kernel(
    float* __restrict__ y) {
  __shared__ float red[256];
  const int row = blockIdx.x;
  const int t = threadIdx.x;
  float* yr = y + (size_t)row * 2048;

  float v[8];
  float m = 0.0f;
#pragma unroll
  for (int i = 0; i < 8; ++i) {
    v[i] = yr[t * 8 + i];
    m = fmaxf(m, fabsf(v[i]));
  }
  red[t] = m;
  __syncthreads();
  for (int s = 128; s > 0; s >>= 1) {
    if (t < s) red[t] = fmaxf(red[t], red[t + s]);
    __syncthreads();
  }
  const float amax = red[0];
  const float scale = amax > 0.0f ? amax * (1.0f / QMAX) : 1.0f;
#pragma unroll
  for (int i = 0; i < 8; ++i) {
    float q = rintf(v[i] / scale);
    q = fminf(fmaxf(q, -QMAX), QMAX);
    yr[t * 8 + i] = q * scale;
  }
}

extern "C" void kernel_launch(void* const* d_in, const int* in_sizes, int n_in,
                              void* d_out, int out_size, void* d_ws,
                              size_t ws_size, hipStream_t stream) {
  (void)in_sizes; (void)n_in; (void)out_size; (void)ws_size;
  const float* x = (const float*)d_in[0];
  const float* kern = (const float*)d_in[1];
  const float* bias = (const float*)d_in[2];
  float* out = (float*)d_out;

  char* ws = (char*)d_ws;
  signed char* qxA = (signed char*)(ws);
  float* sx = (float*)(ws + 16777216);
  signed char* qkB = (signed char*)(ws + 16809984);
  float* sk = (float*)(ws + 21004288);
  float* bq = (float*)(ws + 21012480);

  quant_x_kernel<<<8192, 256, 0, stream>>>(x, qxA, sx);
  quant_w_kernel<<<32, 256, 0, stream>>>(kern, qkB, sk);
  quant_b_kernel<<<1, 256, 0, stream>>>(bias, bq);
  qgemm_wmma_kernel<<<dim3(16, 256), 256, 0, stream>>>(qxA, sx, qkB, sk, bq,
                                                       out);
  requant_out_kernel<<<8192, 256, 0, stream>>>(out);
}
